// MambaBlock_31026843746789
// MI455X (gfx1250) — matrix operand order repaired and checked by reading, not compiled, not run on hardware
//
#include <hip/hip_runtime.h>
#include <hip/hip_bf16.h>

// ---------------------------------------------------------------------------
// Mamba block for MI455X (gfx1250, wave32, WMMA, async-to-LDS double buffer).
//   0) one-time fp32->f16 converts of x and the 4 weight matrices
//   1) in_proj GEMM  (8192x4096x1024)  f16 WMMA -> xz (fp32)
//   2) causal conv(K=4)+SiLU           -> u (fp32) + u_h (f16)
//   3) x_proj GEMM   (8192x96x2048)    -> x_dbl (fp32) + x_dbl_h (f16)
//   4) dt_proj GEMM  (8192x2048x64)    -> dt_raw (fp32)
//   5) selective scan (seq L, parallel B*Di; fused softplus, +u*D, *silu(res))
//      -> y_h (f16)
//   6) out_proj GEMM (8192x1024x2048)  -> d_out (fp32)
// ---------------------------------------------------------------------------

typedef _Float16 f16;
typedef __attribute__((ext_vector_type(16))) _Float16 v16h;
typedef __attribute__((ext_vector_type(8)))  _Float16 v8h;
typedef __attribute__((ext_vector_type(4)))  _Float16 v4h;
typedef __attribute__((ext_vector_type(8)))  float    v8f;
typedef __attribute__((ext_vector_type(4)))  float    v4f;
typedef __attribute__((ext_vector_type(4)))  int      v4i;

union U16h { v16h v; v8h h[2]; };

// Problem constants
#define BB   4
#define LL   2048
#define DM   1024
#define DI   2048
#define NS   16
#define RK   64
#define EE   (2*DI)      // 4096
#define XD   (RK + 2*NS) // 96
#define MM   (BB*LL)     // 8192

// GEMM tiling
#define BM 128
#define BN 128
#define BK 32
#define LDK 40           // padded LDS stride (f16) -> breaks bank conflicts

#if __has_builtin(__builtin_amdgcn_global_load_async_to_lds_b128)
#define HAVE_ASYNC_LDS 1
#endif

typedef __attribute__((address_space(1))) v4i gv4i;
typedef __attribute__((address_space(3))) v4i lv4i;

// one 16-byte global->LDS copy (async DMA if available, else load+store)
__device__ __forceinline__ void copy16_g2l(f16* ldst, const f16* gsrc) {
#ifdef HAVE_ASYNC_LDS
    __builtin_amdgcn_global_load_async_to_lds_b128(
        (gv4i*)gsrc, (lv4i*)ldst, 0, 0);
#else
    *(v8h*)ldst = *(const v8h*)gsrc;
#endif
}

__device__ __forceinline__ void wait_async_4() {
#ifdef HAVE_ASYNC_LDS
# if __has_builtin(__builtin_amdgcn_s_wait_asynccnt)
    __builtin_amdgcn_s_wait_asynccnt(4);
# else
    asm volatile("s_wait_asynccnt 0x4" ::: "memory");
# endif
#endif
}
__device__ __forceinline__ void wait_async_0() {
#ifdef HAVE_ASYNC_LDS
# if __has_builtin(__builtin_amdgcn_s_wait_asynccnt)
    __builtin_amdgcn_s_wait_asynccnt(0);
# else
    asm volatile("s_wait_asynccnt 0x0" ::: "memory");
# endif
#endif
}

// ---------------------------------------------------------------------------
// out[m,n] = sum_k A[m*lda+k] * W[n*ldw+k]   (A: MxK f16 rows, W: NxK f16 rows)
// Requires M % BM == 0, K % BK == 0.  N handled with clamping/guards.
// ---------------------------------------------------------------------------
__global__ __launch_bounds__(256)
void gemm_h_wmma(const f16* __restrict__ A, const f16* __restrict__ W,
                 float* __restrict__ out, f16* __restrict__ out_h,
                 int M, int N, int K, int lda, int ldw, int ldo)
{
    __shared__ f16 sA[2][BM * LDK];
    __shared__ f16 sB[2][BN * LDK];

    const int tid  = threadIdx.x;
    const int lane = tid & 31;
    const int wave = tid >> 5;       // 0..7
    const int wm   = wave >> 2;      // 0..1 : M offset wm*64
    const int wn   = wave & 3;       // 0..3 : N offset wn*32
    const int m0   = blockIdx.y * BM;
    const int n0   = blockIdx.x * BN;
    const int r    = lane & 15;
    const int hh   = lane >> 4;

    v8f acc[4][2];
    #pragma unroll
    for (int i = 0; i < 4; ++i)
        #pragma unroll
        for (int j = 0; j < 2; ++j) acc[i][j] = v8f{};

    // each thread stages two 16B chunks of A and two of B per K-step
    // chunk c (0..511): row = c>>2, col = (c&3)*8
    const int c0row = tid >> 2,         c0col = (tid & 3) << 3;
    const int c1row = (tid + 256) >> 2, c1col = ((tid + 256) & 3) << 3;

    auto stage = [&](int bi, int kb) {
        // A tile: BM x BK (M % BM == 0 -> no guard)
        copy16_g2l(&sA[bi][c0row * LDK + c0col],
                   A + (long)(m0 + c0row) * lda + kb + c0col);
        copy16_g2l(&sA[bi][c1row * LDK + c1col],
                   A + (long)(m0 + c1row) * lda + kb + c1col);
        // B tile: BN x BK; clamp row so every wave issues exactly 4 async ops
        int gn0 = n0 + c0row; if (gn0 >= N) gn0 = N - 1;
        int gn1 = n0 + c1row; if (gn1 >= N) gn1 = N - 1;
        copy16_g2l(&sB[bi][c0row * LDK + c0col],
                   W + (long)gn0 * ldw + kb + c0col);
        copy16_g2l(&sB[bi][c1row * LDK + c1col],
                   W + (long)gn1 * ldw + kb + c1col);
    };

    const int nkt = K / BK;
    stage(0, 0);

    for (int it = 0; it < nkt; ++it) {
        const int cur = it & 1;
        if (it + 1 < nkt) {
            stage(cur ^ 1, (it + 1) * BK);
            wait_async_4();            // current stage's 4 async ops complete
        } else {
            wait_async_0();
        }
        __syncthreads();

        const f16* bufA = sA[cur];
        const f16* bufB = sB[cur];

        // fragments per ISA 7.12.2 (A: lane=row, split-8 K; B: lane=col, split-8 K)
        U16h af[4], bf[2];
        #pragma unroll
        for (int i = 0; i < 4; ++i) {
            const f16* base = bufA + (wm * 64 + i * 16 + r) * LDK;
            af[i].h[0] = *(const v8h*)(base + hh * 8);
            af[i].h[1] = *(const v8h*)(base + 16 + hh * 8);
        }
        #pragma unroll
        for (int j = 0; j < 2; ++j) {
            const f16* base = bufB + (wn * 32 + j * 16 + r) * LDK;
            bf[j].h[0] = *(const v8h*)(base + hh * 8);
            bf[j].h[1] = *(const v8h*)(base + 16 + hh * 8);
        }

        #pragma unroll
        for (int i = 0; i < 4; ++i)
            #pragma unroll
            for (int j = 0; j < 2; ++j)
                acc[i][j] = __builtin_amdgcn_wmma_f32_16x16x32_f16(
                    false, af[i].v, false, bf[j].v,
                    (short)0, acc[i][j], false, false);

        __syncthreads();               // protect buf before it is restaged
    }

    // store: VGPR v -> row (v + hh*8), col = r within each 16x16 tile
    #pragma unroll
    for (int i = 0; i < 4; ++i) {
        #pragma unroll
        for (int j = 0; j < 2; ++j) {
            const int mt = m0 + wm * 64 + i * 16 + hh * 8;
            const int nt = n0 + wn * 32 + j * 16 + r;
            if (nt < N) {
                #pragma unroll
                for (int v = 0; v < 8; ++v)
                    out[(long)(mt + v) * ldo + nt] = acc[i][j][v];
                if (out_h) {
                    #pragma unroll
                    for (int v = 0; v < 8; ++v)
                        out_h[(long)(mt + v) * ldo + nt] = (f16)acc[i][j][v];
                }
            }
        }
    }
}

// ---------------------------------------------------------------------------
// fp32 -> f16 convert, 4 elements/thread
// ---------------------------------------------------------------------------
__global__ __launch_bounds__(256)
void cvt_f32_f16(const float* __restrict__ src, f16* __restrict__ dst, long n4)
{
    const long i = (long)blockIdx.x * blockDim.x + threadIdx.x;
    if (i >= n4) return;
    const v4f v = ((const v4f*)src)[i];
    v4h o; o.x = (f16)v.x; o.y = (f16)v.y; o.z = (f16)v.z; o.w = (f16)v.w;
    ((v4h*)dst)[i] = o;
}

// ---------------------------------------------------------------------------
// Causal depthwise conv1d (K=4) + SiLU.  xz: (B,L,2*Di) fp32 -> u fp32 + f16
// ---------------------------------------------------------------------------
__global__ __launch_bounds__(256)
void conv_silu_kernel(const float* __restrict__ xz,
                      const float* __restrict__ conv_w,
                      const float* __restrict__ conv_b,
                      float* __restrict__ u, f16* __restrict__ u_h)
{
    const long idx = (long)blockIdx.x * blockDim.x + threadIdx.x;
    if (idx >= (long)MM * DI) return;
    const int d   = (int)(idx % DI);
    const long bl = idx / DI;            // b*L + l
    const int l   = (int)(bl % LL);

    float acc = conv_b[d];
    const float w0 = conv_w[d*4 + 0];
    const float w1 = conv_w[d*4 + 1];
    const float w2 = conv_w[d*4 + 2];
    const float w3 = conv_w[d*4 + 3];
    const long base = (bl - l) * EE + d;
    if (l >= 3) acc += xz[base + (long)(l-3)*EE] * w0;
    if (l >= 2) acc += xz[base + (long)(l-2)*EE] * w1;
    if (l >= 1) acc += xz[base + (long)(l-1)*EE] * w2;
    acc += xz[base + (long)l*EE] * w3;

    const float uv = acc / (1.f + __expf(-acc));
    u[idx]   = uv;
    u_h[idx] = (f16)uv;
}

// ---------------------------------------------------------------------------
// Selective scan: one lane per (b,d); h[0..15] in VGPRs; 16 v_exp_f32/step.
// ---------------------------------------------------------------------------
__global__ __launch_bounds__(256)
void scan_kernel(const float* __restrict__ u,
                 const float* __restrict__ x_dbl,
                 const float* __restrict__ dt_raw,
                 const float* __restrict__ dt_b,
                 const float* __restrict__ A_log,
                 const float* __restrict__ Dp,
                 const float* __restrict__ xz,
                 f16* __restrict__ y_h)
{
    const int gid = blockIdx.x * blockDim.x + threadIdx.x;   // 0..B*Di-1
    if (gid >= BB * DI) return;
    const int d = gid % DI;
    const int b = gid / DI;

    float Aa[NS];
    #pragma unroll
    for (int n = 0; n < NS; ++n) Aa[n] = -__expf(A_log[d*NS + n]);

    float h[NS];
    #pragma unroll
    for (int n = 0; n < NS; ++n) h[n] = 0.f;

    const float dtb = dt_b[d];
    const float Dd  = Dp[d];

    for (int t = 0; t < LL; ++t) {
        const long base = (long)b * LL + t;
        const float dtr = dt_raw[base * DI + d] + dtb;
        const float dt  = (dtr > 20.f) ? dtr : log1pf(__expf(dtr)); // softplus
        const float ut  = u[base * DI + d];

        const v4f* xb = (const v4f*)(x_dbl + base * XD + RK);
        const v4f B0 = xb[0], B1 = xb[1], B2 = xb[2], B3 = xb[3];
        const v4f C0 = xb[4], C1 = xb[5], C2 = xb[6], C3 = xb[7];
        float Bv[NS], Cv[NS];
        Bv[0]=B0.x; Bv[1]=B0.y; Bv[2]=B0.z; Bv[3]=B0.w;
        Bv[4]=B1.x; Bv[5]=B1.y; Bv[6]=B1.z; Bv[7]=B1.w;
        Bv[8]=B2.x; Bv[9]=B2.y; Bv[10]=B2.z; Bv[11]=B2.w;
        Bv[12]=B3.x; Bv[13]=B3.y; Bv[14]=B3.z; Bv[15]=B3.w;
        Cv[0]=C0.x; Cv[1]=C0.y; Cv[2]=C0.z; Cv[3]=C0.w;
        Cv[4]=C1.x; Cv[5]=C1.y; Cv[6]=C1.z; Cv[7]=C1.w;
        Cv[8]=C2.x; Cv[9]=C2.y; Cv[10]=C2.z; Cv[11]=C2.w;
        Cv[12]=C3.x; Cv[13]=C3.y; Cv[14]=C3.z; Cv[15]=C3.w;

        const float du = dt * ut;
        float yt = 0.f;
        #pragma unroll
        for (int n = 0; n < NS; ++n) {
            const float dA = __expf(dt * Aa[n]);   // TRANS unit, co-executes
            h[n] = dA * h[n] + du * Bv[n];
            yt  += h[n] * Cv[n];
        }

        const float res  = xz[base * EE + DI + d];
        const float sres = res / (1.f + __expf(-res));
        y_h[base * DI + d] = (f16)((yt + ut * Dd) * sres);
    }
}

// ---------------------------------------------------------------------------
extern "C" void kernel_launch(void* const* d_in, const int* in_sizes, int n_in,
                              void* d_out, int out_size, void* d_ws, size_t ws_size,
                              hipStream_t stream)
{
    const float* x          = (const float*)d_in[0];
    const float* in_proj_w  = (const float*)d_in[1];
    const float* conv_w     = (const float*)d_in[2];
    const float* conv_b     = (const float*)d_in[3];
    const float* x_proj_w   = (const float*)d_in[4];
    const float* dt_proj_w  = (const float*)d_in[5];
    const float* dt_proj_b  = (const float*)d_in[6];
    const float* A_log      = (const float*)d_in[7];
    const float* D_param    = (const float*)d_in[8];
    const float* out_proj_w = (const float*)d_in[9];
    float* out              = (float*)d_out;

    // ---- workspace carve-up (all sizes 256B-aligned) ----
    char* p = (char*)d_ws;
    float* xz     = (float*)p; p += (long)MM * EE * 4;   // 128 MB
    float* u      = (float*)p; p += (long)MM * DI * 4;   //  64 MB
    float* x_dbl  = (float*)p; p += (long)MM * XD * 4;   //   3 MB
    float* dt_raw = (float*)p; p += (long)MM * DI * 4;   //  64 MB
    f16* x_h      = (f16*)p;   p += (long)MM * DM * 2;
    f16* u_h      = (f16*)p;   p += (long)MM * DI * 2;
    f16* xdbl_h   = (f16*)p;   p += (long)MM * XD * 2;
    f16* y_h      = (f16*)p;   p += (long)MM * DI * 2;
    f16* w_in_h   = (f16*)p;   p += (long)EE * DM * 2;
    f16* w_xp_h   = (f16*)p;   p += (long)XD * DI * 2;
    f16* w_dt_h   = (f16*)p;   p += (long)DI * RK * 2;
    f16* w_out_h  = (f16*)p;   p += (long)DM * DI * 2;

    // ---- 0) fp32 -> f16 converts (x + weights) ----
    {
        long n4;
        n4 = (long)MM * DM / 4;
        cvt_f32_f16<<<(unsigned)((n4 + 255) / 256), 256, 0, stream>>>(x, x_h, n4);
        n4 = (long)EE * DM / 4;
        cvt_f32_f16<<<(unsigned)((n4 + 255) / 256), 256, 0, stream>>>(in_proj_w, w_in_h, n4);
        n4 = (long)XD * DI / 4;
        cvt_f32_f16<<<(unsigned)((n4 + 255) / 256), 256, 0, stream>>>(x_proj_w, w_xp_h, n4);
        n4 = (long)DI * RK / 4;
        cvt_f32_f16<<<(unsigned)((n4 + 255) / 256), 256, 0, stream>>>(dt_proj_w, w_dt_h, n4);
        n4 = (long)DM * DI / 4;
        cvt_f32_f16<<<(unsigned)((n4 + 255) / 256), 256, 0, stream>>>(out_proj_w, w_out_h, n4);
    }

    // ---- 1) in_proj: xz = x @ in_proj_w^T  (8192 x 4096 x 1024) ----
    gemm_h_wmma<<<dim3(EE/BN, MM/BM), 256, 0, stream>>>(
        x_h, w_in_h, xz, (f16*)nullptr, MM, EE, DM, DM, DM, EE);

    // ---- 2) conv + silu -> u, u_h ----
    {
        const long n = (long)MM * DI;
        conv_silu_kernel<<<(unsigned)((n + 255) / 256), 256, 0, stream>>>(
            xz, conv_w, conv_b, u, u_h);
    }

    // ---- 3) x_proj: x_dbl = u @ x_proj_w^T  (8192 x 96 x 2048) ----
    gemm_h_wmma<<<dim3((XD + BN - 1)/BN, MM/BM), 256, 0, stream>>>(
        u_h, w_xp_h, x_dbl, xdbl_h, MM, XD, DI, DI, DI, XD);

    // ---- 4) dt_proj: dt_raw = x_dbl[:, :RK] @ dt_proj_w^T  (8192 x 2048 x 64)
    gemm_h_wmma<<<dim3(DI/BN, MM/BM), 256, 0, stream>>>(
        xdbl_h, w_dt_h, dt_raw, (f16*)nullptr, MM, DI, RK, XD, RK, DI);

    // ---- 5) selective scan -> y_h ----
    scan_kernel<<<(BB*DI)/256, 256, 0, stream>>>(
        u, x_dbl, dt_raw, dt_proj_b, A_log, D_param, xz, y_h);

    // ---- 6) out_proj: out = y @ out_proj_w^T  (8192 x 1024 x 2048) ----
    gemm_h_wmma<<<dim3(DM/BN, MM/BM), 256, 0, stream>>>(
        y_h, w_out_h, out, (f16*)nullptr, MM, DM, DI, DI, DI, DM);
}
